// AttentionalSampling_12481174962696
// MI455X (gfx1250) — matrix operand order repaired and checked by reading, not compiled, not run on hardware
//
#include <hip/hip_runtime.h>

typedef _Float16 half_t;
typedef __attribute__((ext_vector_type(16))) _Float16 v16h;
typedef __attribute__((ext_vector_type(8)))  _Float16 v8h;
typedef __attribute__((ext_vector_type(4)))  _Float16 v4h;
typedef __attribute__((ext_vector_type(8)))  float    v8f;
typedef __attribute__((ext_vector_type(4)))  unsigned int v4u;
typedef __attribute__((ext_vector_type(8)))  int v8i;
typedef __attribute__((ext_vector_type(4)))  int v4i;

#define D_MODEL 1152
#define HEADS 8
#define HD 144          // head dim
#define HDP 160         // head dim padded to 5*32 WMMA K-steps
#define HW 729
#define HWP 768         // padded to 12 column-tiles of 64
#define NTILE (HWP / 64)
#define MTRK 512
#define TFRAMES 64
#define MQ (TFRAMES * MTRK)          // 32768 (multiple of 128)
#define MK (TFRAMES * HW)            // 46656
#define MKP 46720                    // padded to multiple of 128

__device__ inline v8f wmma_f16(v16h a, v16h b, v8f c) {
  return __builtin_amdgcn_wmma_f32_16x16x32_f16(false, a, false, b, (short)0, c,
                                                false, false);
}

__device__ inline v8f vzero8() {
  v8f r;
#pragma unroll
  for (int i = 0; i < 8; ++i) r[i] = 0.f;
  return r;
}

// Build a v16h A/B fragment from two contiguous 8-half (16B) chunks.
__device__ inline v16h ld16_2(const half_t* plo, const half_t* phi) {
  v8h lo = *(const v8h*)plo;
  v8h hi = *(const v8h*)phi;
  v16h r;
#pragma unroll
  for (int i = 0; i < 8; ++i) { r[i] = lo[i]; r[i + 8] = hi[i]; }
  return r;
}

// ---------------------------------------------------------------------------
// TDM: DMA a 2D tile (rows x dim0_units of 8B) from global into LDS.
// Pad config: after every 16 DWORDs (64B) insert 4 DWORDs (16B) -> a 64B
// data chunk occupies 80B of LDS, i.e. a 32-half chunk strides 40 halfs.
// D# layout per CDNA5 ISA ch.8.3/8.4; groups 2/3 (and trailing group) zero
// for a 2D tensor. This toolchain uses the 6-arg builtin form.
// ---------------------------------------------------------------------------
__device__ inline void tdm_load_2d(unsigned lds_addr, const void* gaddr,
                                   unsigned dim0_units, unsigned rows,
                                   unsigned stride_units) {
  unsigned long long ga = (unsigned long long)(uintptr_t)gaddr;
  v4u g0 = {1u,                                   // count=1, user descriptor
            lds_addr,                             // LDS byte address
            (unsigned)ga,                         // global_addr[31:0]
            (unsigned)((ga >> 32) & 0x1ffffffu) | (2u << 30)};  // type=2
  v8i g1 = {(int)((3u << 16) |                    // data_size = 8B
                  (1u << 20) |                    // pad_enable
                  (3u << 22) |                    // pad_interval: 16 DWORDs
                  (3u << 25)),                    // pad_amount: 4 DWORDs
            (int)(dim0_units << 16),              // tensor_dim0[15:0]
            (int)(rows << 16),                    // tensor_dim1[15:0]
            (int)(dim0_units << 16),              // tile_dim0
            (int)rows,                            // tile_dim1
            (int)stride_units,                    // tensor_dim0_stride[31:0]
            0, 0};
  v4i gz4 = {0, 0, 0, 0};
  v8i gz8 = {0, 0, 0, 0, 0, 0, 0, 0};
  __builtin_amdgcn_tensor_load_to_lds(g0, g1, gz4, gz4, gz8, 0);
}

// ---------------------------------------------------------------------------
// f32 -> f16 convert (vector-of-4), zero-fill beyond n4valid (row padding).
// ---------------------------------------------------------------------------
__global__ __launch_bounds__(256) void cvt_f16_pad(const float* __restrict__ src,
                                                   half_t* __restrict__ dst,
                                                   size_t n4valid, size_t n4total) {
  size_t i = (size_t)blockIdx.x * 256 + threadIdx.x;
  if (i >= n4total) return;
  v4h o;
  if (i < n4valid) {
    float4 v = *(const float4*)(src + i * 4);
    o[0] = (half_t)v.x; o[1] = (half_t)v.y;
    o[2] = (half_t)v.z; o[3] = (half_t)v.w;
  } else {
    o[0] = o[1] = o[2] = o[3] = (half_t)0.f;
  }
  *(v4h*)(dst + i * 4) = o;
}

// ---------------------------------------------------------------------------
// W [K][N] f32  ->  Wt [N][K] f16  (LDS-tiled transpose + convert)
// ---------------------------------------------------------------------------
__global__ __launch_bounds__(256) void transpose_cvt(const float* __restrict__ W,
                                                     half_t* __restrict__ Wt,
                                                     int dim) {
  __shared__ float tile[32][33];
  const int bx = blockIdx.x * 32, by = blockIdx.y * 32;
  for (int i = threadIdx.y; i < 32; i += 8)
    tile[i][threadIdx.x] = W[(size_t)(by + i) * dim + bx + threadIdx.x];
  __syncthreads();
  for (int i = threadIdx.y; i < 32; i += 8)
    Wt[(size_t)(bx + i) * dim + by + threadIdx.x] = (half_t)tile[threadIdx.x][i];
}

// ---------------------------------------------------------------------------
// C[M,N](f32) = A[M,K](f16) @ Bt[N,K](f16 = B^T). Block tile 128x128,
// K-tile 32, TDM-fed double-buffered LDS, 8 waves (4Mx2N), 2x4 WMMA tiles
// per wave. A rows must be padded to 128. K-tile staging is done entirely
// by the Tensor Data Mover (wave 0 issues; TENSORcnt + barrier publishes).
// ---------------------------------------------------------------------------
__global__ __launch_bounds__(256) void gemm_wmma_f16(
    const half_t* __restrict__ A, const half_t* __restrict__ Bt,
    float* __restrict__ C, int N, int K) {
  __shared__ alignas(16) half_t As[2][128 * 40];
  __shared__ alignas(16) half_t Bs[2][128 * 40];

  const int tid = threadIdx.x;
  const int lane = tid & 31;
  const int wid = tid >> 5;
  const int lo16 = lane & 15;
  const int hi = lane >> 4;
  const int m0 = blockIdx.y * 128;
  const int n0 = blockIdx.x * 128;
  const int wm = wid & 3;
  const int wn = wid >> 2;
  const unsigned su = (unsigned)(K / 4);  // row stride in 8B units (K halfs)

  v8f acc[2][4];
#pragma unroll
  for (int i = 0; i < 2; ++i)
#pragma unroll
    for (int j = 0; j < 4; ++j) acc[i][j] = vzero8();

  if (wid == 0) {
    tdm_load_2d((unsigned)(uintptr_t)&As[0][0], A + (size_t)m0 * K, 8, 128, su);
    tdm_load_2d((unsigned)(uintptr_t)&Bs[0][0], Bt + (size_t)n0 * K, 8, 128, su);
  }

  const int nk = K / 32;
  for (int it = 0; it < nk; ++it) {
    const int cur = it & 1;
    if (wid == 0) __builtin_amdgcn_s_wait_tensorcnt(0);
    __syncthreads();  // tile `it` ready; everyone done with buffer `1-cur`
    if (wid == 0 && it + 1 < nk) {
      const int k1 = (it + 1) * 32;
      tdm_load_2d((unsigned)(uintptr_t)&As[1 - cur][0],
                  A + (size_t)m0 * K + k1, 8, 128, su);
      tdm_load_2d((unsigned)(uintptr_t)&Bs[1 - cur][0],
                  Bt + (size_t)n0 * K + k1, 8, 128, su);
    }

    const half_t* as = As[cur];
    const half_t* bs = Bs[cur];
    v16h a[2], b[4];
#pragma unroll
    for (int ms = 0; ms < 2; ++ms) {
      int row = wm * 32 + ms * 16 + lo16;
      int kb = hi * 8;  // A frag: K {kb..kb+8, 16+kb..}
      a[ms] = ld16_2(&as[row * 40 + kb], &as[row * 40 + 16 + kb]);
    }
#pragma unroll
    for (int ns = 0; ns < 4; ++ns) {
      int col = wn * 64 + ns * 16 + lo16;
      int kk = hi * 8;  // B frag: K {kk..kk+8, 16+kk..}
      b[ns] = ld16_2(&bs[col * 40 + kk], &bs[col * 40 + 16 + kk]);
    }
#pragma unroll
    for (int ms = 0; ms < 2; ++ms)
#pragma unroll
      for (int ns = 0; ns < 4; ++ns)
        acc[ms][ns] = wmma_f16(a[ms], b[ns], acc[ms][ns]);
  }

#pragma unroll
  for (int ms = 0; ms < 2; ++ms)
#pragma unroll
    for (int ns = 0; ns < 4; ++ns)
#pragma unroll
      for (int j = 0; j < 8; ++j) {
        int row = m0 + wm * 32 + ms * 16 + j + 8 * hi;
        int col = n0 + wn * 64 + ns * 16 + lo16;
        C[(size_t)row * N + col] = acc[ms][ns][j];
      }
}

// ---------------------------------------------------------------------------
// LayerNorm(Q) and repack to f16 [T][H][512][160] (zero-padded head dim).
// ---------------------------------------------------------------------------
__global__ __launch_bounds__(256) void post_q(const float* __restrict__ Qf,
                                              const float* __restrict__ w,
                                              half_t* __restrict__ Qh) {
  const int r = blockIdx.x * 8 + (threadIdx.x >> 5);
  const int lane = threadIdx.x & 31;
  const float* x = Qf + (size_t)r * D_MODEL;
  float vals[36], s = 0.f, ss = 0.f;
#pragma unroll
  for (int i = 0; i < 36; ++i) {
    float v = x[lane + i * 32];
    vals[i] = v; s += v; ss += v * v;
  }
#pragma unroll
  for (int off = 16; off > 0; off >>= 1) {
    s += __shfl_xor(s, off, 32);
    ss += __shfl_xor(ss, off, 32);
  }
  float mu = s * (1.f / D_MODEL);
  float rstd = rsqrtf(ss * (1.f / D_MODEL) - mu * mu + 1e-6f);
  const int t = r >> 9, m = r & 511;
#pragma unroll
  for (int i = 0; i < 36; ++i) {
    int e = lane + i * 32;
    int h = e / HD, d = e % HD;
    Qh[(((size_t)t * HEADS + h) * MTRK + m) * HDP + d] =
        (half_t)((vals[i] - mu) * rstd * w[e]);
  }
  if (lane < 16)
#pragma unroll
    for (int h = 0; h < HEADS; ++h)
      Qh[(((size_t)t * HEADS + h) * MTRK + m) * HDP + HD + lane] = (half_t)0.f;
}

// ---------------------------------------------------------------------------
// RoPE-2D + LayerNorm on K -> f16 [T][H][768][160]; V (= features f16)
// copied to [T][H][768][144]. Padded rows (n>=729) zeroed.
// ---------------------------------------------------------------------------
__global__ __launch_bounds__(256) void post_k(const float* __restrict__ Kf,
                                              const half_t* __restrict__ featH,
                                              const float* __restrict__ fpos,
                                              const float* __restrict__ kw,
                                              half_t* __restrict__ Kh,
                                              half_t* __restrict__ Vh) {
  const int r = blockIdx.x * 8 + (threadIdx.x >> 5);  // [0, 64*768)
  const int lane = threadIdx.x & 31;
  const int t = r / HWP, n = r % HWP;
  const bool valid = (n < HW);

  if (!valid) {
#pragma unroll
    for (int i = 0; i < 36; ++i) {
      int e = lane + i * 32;
      int h = e / HD, d = e % HD;
      Kh[(((size_t)t * HEADS + h) * HWP + n) * HDP + d] = (half_t)0.f;
      Vh[(((size_t)t * HEADS + h) * HWP + n) * HD + d] = (half_t)0.f;
    }
    if (lane < 16)
#pragma unroll
      for (int h = 0; h < HEADS; ++h)
        Kh[(((size_t)t * HEADS + h) * HWP + n) * HDP + HD + lane] = (half_t)0.f;
    return;
  }

  const float* x = Kf + ((size_t)t * HW + n) * D_MODEL;
  const float px = fpos[n * 2 + 0];
  const float py = fpos[n * 2 + 1];
  float v[36], s = 0.f, ss = 0.f;
#pragma unroll
  for (int j = 0; j < 9; ++j) {
    int p = lane + j * 32;  // rotation pair index 0..287
    float th = __powf(10000.f, -(float)p * (1.f / 288.f));
    float sn, cs;
    __sincosf(px * th, &sn, &cs);
    float a0 = x[2 * p], a1 = x[2 * p + 1];
    v[2 * j + 0] = a0 * cs - a1 * sn;
    v[2 * j + 1] = a0 * sn + a1 * cs;
    __sincosf(py * th, &sn, &cs);
    float b0 = x[576 + 2 * p], b1 = x[577 + 2 * p];
    v[18 + 2 * j + 0] = b0 * cs - b1 * sn;
    v[18 + 2 * j + 1] = b0 * sn + b1 * cs;
  }
#pragma unroll
  for (int i = 0; i < 36; ++i) { s += v[i]; ss += v[i] * v[i]; }
#pragma unroll
  for (int off = 16; off > 0; off >>= 1) {
    s += __shfl_xor(s, off, 32);
    ss += __shfl_xor(ss, off, 32);
  }
  float mu = s * (1.f / D_MODEL);
  float rstd = rsqrtf(ss * (1.f / D_MODEL) - mu * mu + 1e-6f);

  auto wr = [&](int e, float val) {
    int h = e / HD, d = e % HD;
    Kh[(((size_t)t * HEADS + h) * HWP + n) * HDP + d] =
        (half_t)((val - mu) * rstd * kw[e]);
  };
#pragma unroll
  for (int j = 0; j < 9; ++j) {
    int p = lane + j * 32;
    wr(2 * p, v[2 * j]);            wr(2 * p + 1, v[2 * j + 1]);
    wr(576 + 2 * p, v[18 + 2 * j]); wr(577 + 2 * p, v[18 + 2 * j + 1]);
  }
  if (lane < 16)
#pragma unroll
    for (int h = 0; h < HEADS; ++h)
      Kh[(((size_t)t * HEADS + h) * HWP + n) * HDP + HD + lane] = (half_t)0.f;

  const half_t* fr = featH + ((size_t)t * HW + n) * D_MODEL;
#pragma unroll
  for (int i = 0; i < 36; ++i) {
    int e = lane + i * 32;
    int h = e / HD, d = e % HD;
    Vh[(((size_t)t * HEADS + h) * HWP + n) * HD + d] = fr[e];
  }
}

// ---------------------------------------------------------------------------
// Flash-style attention per (t,h): 8 waves, 128 query rows per block,
// 12 column tiles of 64. K tiles arrive via TDM (320B rows -> 400B LDS
// stride with the 64B+16B pad pattern); V tiles are prefetched manually
// into a second double buffer. One barrier per tile.
// ---------------------------------------------------------------------------
__global__ __launch_bounds__(256) void attn_kernel(
    const half_t* __restrict__ Qh, const half_t* __restrict__ Kh,
    const half_t* __restrict__ Vh, const float* __restrict__ tracks,
    const float* __restrict__ fpos, half_t* __restrict__ Out) {
  __shared__ float s_trk[256];                      // 128 rows x (x,y)
  __shared__ float s_fp[HW * 2];                    // feature positions
  __shared__ alignas(16) half_t s_K[2][64 * 200];   // TDM tile, 200-half rows
  __shared__ alignas(16) half_t s_V[2][HD * 72];    // [dv][n] 64 + pad 8
  __shared__ alignas(16) half_t s_P[8 * 16 * 64];   // per-wave P tiles

  const int tid = threadIdx.x;
  const int lane = tid & 31;
  const int wid = tid >> 5;
  const int lo16 = lane & 15;
  const int hi = lane >> 4;
  const int th = blockIdx.y;
  const int t = th >> 3, h = th & 7;
  const int m0 = blockIdx.x * 128;

  for (int i = tid; i < HW * 2; i += 256) s_fp[i] = fpos[i];
  s_trk[tid] = tracks[((size_t)t * MTRK + m0) * 2 + tid];

  // Q fragments for this wave's 16 rows, 5 WMMA K-steps (HDP=160)
  v16h qf[5];
  {
    size_t qb = (((size_t)t * HEADS + h) * MTRK + m0 + wid * 16 + lo16) * HDP;
    int kb = hi * 8;
#pragma unroll
    for (int ks = 0; ks < 5; ++ks)
      qf[ks] = ld16_2(Qh + qb + ks * 32 + kb, Qh + qb + ks * 32 + 16 + kb);
  }

  v8f Oacc[9];
#pragma unroll
  for (int d = 0; d < 9; ++d) Oacc[d] = vzero8();
  float m_run[8], l_run[8];
#pragma unroll
  for (int j = 0; j < 8; ++j) { m_run[j] = -3e38f; l_run[j] = 0.f; }

  const size_t kvbase = ((size_t)t * HEADS + h) * HWP;

  auto stageV = [&](int n0, half_t* vb) {
#pragma unroll
    for (int i = 0; i < 5; ++i) {
      int cq = tid + i * 256;
      if (cq < 1152) {  // 64 rows x 72 halfs (dv 0..71) as b64 chunks
        int rr = cq / 18, c4 = (cq % 18) * 4;
        v4h v = *(const v4h*)(Vh + (kvbase + n0 + rr) * HD + c4);
#pragma unroll
        for (int j = 0; j < 4; ++j) vb[(c4 + j) * 72 + rr] = v[j];
      }
    }
  };

  // Prologue: V tile 0 manual, K tile 0 via TDM
  stageV(0, &s_V[0][0]);
  if (wid == 0)
    tdm_load_2d((unsigned)(uintptr_t)&s_K[0][0], Kh + kvbase * HDP, 40, 64, 40);

  for (int nt = 0; nt < NTILE; ++nt) {
    const int cur = nt & 1;
    if (wid == 0) __builtin_amdgcn_s_wait_tensorcnt(0);
    __syncthreads();  // K(nt)/V(nt) visible; buffers 1-cur free
    if (nt + 1 < NTILE) {
      if (wid == 0)
        tdm_load_2d((unsigned)(uintptr_t)&s_K[1 - cur][0],
                    Kh + (kvbase + (size_t)(nt + 1) * 64) * HDP, 40, 64, 40);
      stageV((nt + 1) * 64, &s_V[1 - cur][0]);
    }
    const half_t* sk = &s_K[cur][0];
    const half_t* sv2 = &s_V[cur][0];

    // S = Q @ K^T (scaled) + gaussian bias
    float sv[4][8];
#pragma unroll
    for (int ns = 0; ns < 4; ++ns) {
      v8f acc = vzero8();
      int col = ns * 16 + lo16;
      const half_t* bp = sk + col * 200 + hi * 8;  // padded TDM layout
#pragma unroll
      for (int ks = 0; ks < 5; ++ks)
        acc = wmma_f16(qf[ks], ld16_2(bp + ks * 40, bp + ks * 40 + 16), acc);
      int gn = nt * 64 + col;
      bool inb = gn < HW;
      float fx = 0.f, fy = 0.f;
      if (inb) { fx = s_fp[gn * 2]; fy = s_fp[gn * 2 + 1]; }
#pragma unroll
      for (int j = 0; j < 8; ++j) {
        int mrel = wid * 16 + j + 8 * hi;
        float bias = -3e38f;
        if (inb) {
          float dx = s_trk[mrel * 2] - fx, dy = s_trk[mrel * 2 + 1] - fy;
          bias = -(dx * dx + dy * dy) * 0.125f;  // 1/(2*sigma^2), sigma=2
        }
        sv[ns][j] = acc[j] * (1.f / 12.f) + bias;  // 1/sqrt(144)
      }
    }

    // Online softmax (rows live in half-wave groups; width-16 shuffles)
    float alpha[8], tsum[8];
#pragma unroll
    for (int j = 0; j < 8; ++j) {
      float tm = fmaxf(fmaxf(sv[0][j], sv[1][j]), fmaxf(sv[2][j], sv[3][j]));
#pragma unroll
      for (int off = 1; off < 16; off <<= 1) tm = fmaxf(tm, __shfl_xor(tm, off, 16));
      float mn = fmaxf(m_run[j], tm);
      alpha[j] = __expf(m_run[j] - mn);
      m_run[j] = mn;
      tsum[j] = 0.f;
    }
    half_t* pw = &s_P[wid * 16 * 64];
#pragma unroll
    for (int ns = 0; ns < 4; ++ns)
#pragma unroll
      for (int j = 0; j < 8; ++j) {
        float p = __expf(sv[ns][j] - m_run[j]);
        tsum[j] += p;
        pw[(j + 8 * hi) * 64 + ns * 16 + lo16] = (half_t)p;
      }
#pragma unroll
    for (int j = 0; j < 8; ++j) {
      float tsj = tsum[j];
#pragma unroll
      for (int off = 1; off < 16; off <<= 1) tsj += __shfl_xor(tsj, off, 16);
      l_run[j] = l_run[j] * alpha[j] + tsj;
#pragma unroll
      for (int d = 0; d < 9; ++d) Oacc[d][j] *= alpha[j];
    }
    // Intra-wave LDS RAW hazard: P stored in C-layout, re-read in A-layout.
    asm volatile("s_wait_dscnt 0x0" ::: "memory");

    // O += P @ V  (K = 64 -> 2 WMMA K-steps)
#pragma unroll
    for (int ks2 = 0; ks2 < 2; ++ks2) {
      const half_t* pa = pw + lo16 * 64 + ks2 * 32 + hi * 8;
      v16h a = ld16_2(pa, pa + 16);
#pragma unroll
      for (int d = 0; d < 9; ++d) {
        const half_t* bp = sv2 + (d * 16 + lo16) * 72 + ks2 * 32 + hi * 8;
        Oacc[d] = wmma_f16(a, ld16_2(bp, bp + 16), Oacc[d]);
      }
    }
  }

  // Normalize; write sampled as f16 [T][M][1152] (head-concatenated)
  const int mbase = m0 + wid * 16;
#pragma unroll
  for (int j = 0; j < 8; ++j) {
    float inv = 1.f / l_run[j];
    int row = mbase + j + 8 * hi;
    half_t* orow = Out + ((size_t)t * MTRK + row) * D_MODEL + h * HD;
#pragma unroll
    for (int d = 0; d < 9; ++d)
      orow[d * 16 + lo16] = (half_t)(Oacc[d][j] * inv);
  }
}

// ---------------------------------------------------------------------------
extern "C" void kernel_launch(void* const* d_in, const int* in_sizes, int n_in,
                              void* d_out, int out_size, void* d_ws,
                              size_t ws_size, hipStream_t stream) {
  (void)in_sizes; (void)n_in; (void)out_size; (void)ws_size;
  const float* features     = (const float*)d_in[0];
  const float* tracks       = (const float*)d_in[1];
  const float* track_tokens = (const float*)d_in[2];
  const float* fpos         = (const float*)d_in[3];
  const float* W_q          = (const float*)d_in[4];
  const float* W_k          = (const float*)d_in[5];
  const float* qnw          = (const float*)d_in[6];
  const float* knw          = (const float*)d_in[7];
  const float* Wo           = (const float*)d_in[8];

  size_t off = 0;
  auto carve = [&](size_t bytes) {
    size_t cur = off;
    off += (bytes + 255) & ~(size_t)255;
    return (char*)d_ws + cur;
  };
  half_t* Aq  = (half_t*)carve((size_t)MQ * D_MODEL * 2);    // track_tokens f16
  half_t* Ak  = (half_t*)carve((size_t)MKP * D_MODEL * 2);   // features f16 (padded)
  half_t* Wqt = (half_t*)carve((size_t)D_MODEL * D_MODEL * 2);
  half_t* Wkt = (half_t*)carve((size_t)D_MODEL * D_MODEL * 2);
  half_t* Wot = (half_t*)carve((size_t)D_MODEL * D_MODEL * 2);
  float*  Qf  = (float*)carve((size_t)MQ * D_MODEL * 4);
  float*  Kf  = (float*)carve((size_t)MKP * D_MODEL * 4);
  half_t* Qh  = (half_t*)carve((size_t)TFRAMES * HEADS * MTRK * HDP * 2);
  half_t* Kh  = (half_t*)carve((size_t)TFRAMES * HEADS * HWP * HDP * 2);
  half_t* Vh  = (half_t*)carve((size_t)TFRAMES * HEADS * HWP * HD * 2);
  half_t* sampled = Aq;  // reuse: Aq is consumed by the Q GEMM before attn

  // ---- prep: f16 conversions + weight transposes ----
  {
    size_t n4q = (size_t)MQ * D_MODEL / 4;
    cvt_f16_pad<<<(unsigned)((n4q + 255) / 256), 256, 0, stream>>>(
        track_tokens, Aq, n4q, n4q);
    size_t n4kv = (size_t)MK * D_MODEL / 4;
    size_t n4kt = (size_t)MKP * D_MODEL / 4;
    cvt_f16_pad<<<(unsigned)((n4kt + 255) / 256), 256, 0, stream>>>(
        features, Ak, n4kv, n4kt);
    dim3 tb(32, 8), tg(D_MODEL / 32, D_MODEL / 32);
    transpose_cvt<<<tg, tb, 0, stream>>>(W_q, Wqt, D_MODEL);
    transpose_cvt<<<tg, tb, 0, stream>>>(W_k, Wkt, D_MODEL);
    transpose_cvt<<<tg, tb, 0, stream>>>(Wo, Wot, D_MODEL);
  }

  // ---- projections ----
  dim3 gq(D_MODEL / 128, MQ / 128);   // 9 x 256
  dim3 gk(D_MODEL / 128, MKP / 128);  // 9 x 365
  gemm_wmma_f16<<<gq, 256, 0, stream>>>(Aq, Wqt, Qf, D_MODEL, D_MODEL);
  gemm_wmma_f16<<<gk, 256, 0, stream>>>(Ak, Wkt, Kf, D_MODEL, D_MODEL);

  // ---- LN / RoPE / repack ----
  post_q<<<MQ / 8, 256, 0, stream>>>(Qf, qnw, Qh);
  post_k<<<(TFRAMES * HWP) / 8, 256, 0, stream>>>(Kf, Ak, fpos, knw, Kh, Vh);

  // ---- attention ----
  dim3 ga(MTRK / 128, TFRAMES * HEADS);  // 4 x 512
  attn_kernel<<<ga, 256, 0, stream>>>(Qh, Kh, Vh, tracks, fpos, sampled);

  // ---- output projection ----
  gemm_wmma_f16<<<gq, 256, 0, stream>>>(sampled, Wot, (float*)d_out,
                                        D_MODEL, D_MODEL);
}
